// GraphT5_GINEncoder_12163347383180
// MI455X (gfx1250) — compile-verified, matrix operand order checked by reading, not run on hardware
//
#include <hip/hip_runtime.h>
#include <hip/hip_bf16.h>
#include <cstdint>
#include <cstddef>

typedef __attribute__((ext_vector_type(16))) _Float16 v16h;
typedef __attribute__((ext_vector_type(8)))  _Float16 v8h;
typedef __attribute__((ext_vector_type(8)))  float    v8f;

// ---------------- problem constants ----------------
constexpr int kN   = 32768;   // nodes
constexpr int kE   = 131072;  // edges
constexpr int kH   = 512;
constexpr int kHE  = 128;
constexpr int kOUT = 1024;
constexpr int kL   = 6;
constexpr int kB   = 1024;    // graphs
constexpr int kNPG = 32;      // nodes per graph

__constant__ int NODE_OFF_D[9] = {0,119,128,139,151,160,165,173,175};
__constant__ int EDGE_OFF_D[3] = {0,119,128};

// Assemble a 16-bit 16x32 WMMA fragment from two contiguous 16-byte loads.
// Per-lane layout: lane = r + 16*khalf;
//   halves[0..7]  = row r, K = khalf*8 .. +8
//   halves[8..15] = row r, K = 16 + khalf*8 .. +8
__device__ __forceinline__ v16h frag_from(const _Float16* __restrict__ p) {
  v8h lo = *(const v8h*)p;
  v8h hi = *(const v8h*)(p + 16);
  v16h f;
#pragma unroll
  for (int i = 0; i < 8; ++i) { f[i] = lo[i]; f[i + 8] = hi[i]; }
  return f;
}

// ---------------- WMMA GEMM: C[M,N] = A[M,K] * Bt[N,K]^T + bias ----------------
// 256 threads = 8 waves; each wave computes a 16(M) x 64(N) strip (4 tiles).
// Per K-step of 32: issue all 10 b128 fragment loads as one batch, then run
// 4 back-to-back WMMAs behind partial loadcnt waits. All fragment pointers
// are named scalars derived straight from the kernel args so the address-
// space inference keeps every load on the global (LOADcnt-only) path.
__global__ __launch_bounds__(256)
void gemm_wmma_f16(const _Float16* __restrict__ A,
                   const _Float16* __restrict__ Bt,
                   const float* __restrict__ bias,
                   float* __restrict__ C,
                   int M, int N, int K) {
  const int lane = threadIdx.x & 31;
  const int wave = (int)(blockIdx.x * (blockDim.x >> 5)) + (threadIdx.x >> 5);
  const int ngrp = N >> 6;                  // N / 64
  const int mt   = wave / ngrp;
  const int ng   = wave - mt * ngrp;
  if (mt * 16 >= M) return;                 // wave-uniform: EXEC stays all-ones
  const int m0 = mt * 16;
  const int n0 = ng * 64;

  const int r     = lane & 15;
  const int khalf = lane >> 4;
  const int koff  = khalf * 8;

  // Per-lane fragment pointers, stepped by 32 halves (64B) per K iteration.
  const _Float16* pa  = A  + (size_t)(m0 + r) * K + koff;
  const _Float16* pb0 = Bt + (size_t)(n0 +  0 + r) * K + koff;
  const _Float16* pb1 = Bt + (size_t)(n0 + 16 + r) * K + koff;
  const _Float16* pb2 = Bt + (size_t)(n0 + 32 + r) * K + koff;
  const _Float16* pb3 = Bt + (size_t)(n0 + 48 + r) * K + koff;

  v8f acc0 = {}, acc1 = {}, acc2 = {}, acc3 = {};

  for (int k0 = 0; k0 < K; k0 += 32) {
    // Batch all fragment loads (clause-friendly), then compute.
    v16h a  = frag_from(pa);
    v16h b0 = frag_from(pb0);
    v16h b1 = frag_from(pb1);
    v16h b2 = frag_from(pb2);
    v16h b3 = frag_from(pb3);
    if (k0 + 32 < K)
      __builtin_prefetch((const void*)(pa + 32), 0, 1);   // global_prefetch_b8
    acc0 = __builtin_amdgcn_wmma_f32_16x16x32_f16(false, a, false, b0, (short)0, acc0, false, false);
    acc1 = __builtin_amdgcn_wmma_f32_16x16x32_f16(false, a, false, b1, (short)0, acc1, false, false);
    acc2 = __builtin_amdgcn_wmma_f32_16x16x32_f16(false, a, false, b2, (short)0, acc2, false, false);
    acc3 = __builtin_amdgcn_wmma_f32_16x16x32_f16(false, a, false, b3, (short)0, acc3, false, false);
    pa += 32; pb0 += 32; pb1 += 32; pb2 += 32; pb3 += 32;
  }

  // C/D layout: VGPR rr -> row (rr + 8*(lane>>4)), col = lane&15
  const int cn = lane & 15;
  const int mh = lane >> 4;
  v8f accs[4] = {acc0, acc1, acc2, acc3};
#pragma unroll
  for (int j = 0; j < 4; ++j) {
    const float bv = bias ? bias[n0 + 16 * j + cn] : 0.f;
#pragma unroll
    for (int rr = 0; rr < 8; ++rr)
      C[(size_t)(m0 + rr + 8 * mh) * N + (n0 + 16 * j + cn)] = accs[j][rr] + bv;
  }
}

// ---------------- elementwise / embedding kernels ----------------
__global__ void node_embed_kernel(const int* __restrict__ x, const float* __restrict__ tab,
                                  float* __restrict__ z, _Float16* __restrict__ zh, size_t total) {
  size_t i = (size_t)blockIdx.x * blockDim.x + threadIdx.x;
  if (i >= total) return;
  const size_t n = i >> 9;          // / 512
  const int    h = (int)(i & 511);
  float s = 0.f;
#pragma unroll
  for (int f = 0; f < 9; ++f)
    s += tab[(size_t)(x[n * 9 + f] + NODE_OFF_D[f]) * kH + h];
  z[i]  = s;
  zh[i] = (_Float16)s;
}

__global__ void edge_embed_kernel(const int* __restrict__ ea, const float* __restrict__ tab,
                                  _Float16* __restrict__ eh, size_t total) {
  size_t i = (size_t)blockIdx.x * blockDim.x + threadIdx.x;
  if (i >= total) return;
  const size_t e = i >> 7;          // / 128
  const int    h = (int)(i & 127);
  float s = 0.f;
#pragma unroll
  for (int f = 0; f < 3; ++f)
    s += tab[(size_t)(ea[e * 3 + f] + EDGE_OFF_D[f]) * kHE + h];
  eh[i] = (_Float16)s;
}

__global__ void transpose_cvt_kernel(const float* __restrict__ W, _Float16* __restrict__ Wt,
                                     int K, int N) {  // W[K][N] -> Wt[N][K] (f16)
  size_t i = (size_t)blockIdx.x * blockDim.x + threadIdx.x;
  const size_t total = (size_t)K * N;
  if (i >= total) return;
  const int k = (int)(i / N);
  const int n = (int)(i - (size_t)k * N);
  Wt[(size_t)n * K + k] = (_Float16)W[i];
}

__global__ void zero_f32_kernel(float* __restrict__ p, size_t n) {
  size_t i = (size_t)blockIdx.x * blockDim.x + threadIdx.x;
  if (i < n) p[i] = 0.f;
}

__global__ void cvt_f16_kernel(const float* __restrict__ a, _Float16* __restrict__ o, size_t n) {
  size_t i = (size_t)blockIdx.x * blockDim.x + threadIdx.x;
  if (i < n) o[i] = (_Float16)a[i];
}

__global__ void add_cvt_kernel(const float* __restrict__ a, const float* __restrict__ b,
                               _Float16* __restrict__ o, size_t n) {
  size_t i = (size_t)blockIdx.x * blockDim.x + threadIdx.x;
  if (i < n) o[i] = (_Float16)(a[i] + b[i]);
}

// ---------------- BatchNorm (training-mode, column stats) ----------------
__global__ void bn_partial_kernel(const float* __restrict__ X, int M, int N,
                                  float* __restrict__ sums, float* __restrict__ sqs) {
  const int c   = threadIdx.x & 63;
  const int s   = threadIdx.x >> 6;               // 0..3
  const int col = blockIdx.x * 64 + c;
  const int rows = (M + gridDim.y - 1) / gridDim.y;
  const int r0 = blockIdx.y * rows;
  const int r1 = min(M, r0 + rows);
  float sum = 0.f, sq = 0.f;
  for (int i = r0 + s; i < r1; i += 4) {
    const float v = X[(size_t)i * N + col];
    sum += v; sq += v * v;
  }
  __shared__ float sh[2][4][64];
  sh[0][s][c] = sum; sh[1][s][c] = sq;
  __syncthreads();
  if (s == 0) {
    atomicAdd(&sums[col], sh[0][0][c] + sh[0][1][c] + sh[0][2][c] + sh[0][3][c]);
    atomicAdd(&sqs[col],  sh[1][0][c] + sh[1][1][c] + sh[1][2][c] + sh[1][3][c]);
  }
}

__global__ void bn_finalize_kernel(const float* __restrict__ sums, const float* __restrict__ sqs,
                                   float* __restrict__ mean, float* __restrict__ rstd,
                                   int M, int N) {
  int j = blockIdx.x * blockDim.x + threadIdx.x;
  if (j >= N) return;
  const float m   = sums[j] / (float)M;
  const float var = sqs[j] / (float)M - m * m;
  mean[j] = m;
  rstd[j] = rsqrtf(var + 1e-5f);
}

__global__ void bn_relu_apply_kernel(const float* __restrict__ X,
                                     const float* __restrict__ mean, const float* __restrict__ rstd,
                                     const float* __restrict__ g, const float* __restrict__ b,
                                     float* __restrict__ Yf, _Float16* __restrict__ Yh,
                                     size_t total, int N) {
  size_t i = (size_t)blockIdx.x * blockDim.x + threadIdx.x;
  if (i >= total) return;
  const int j = (int)(i % (size_t)N);
  float v = (X[i] - mean[j]) * rstd[j] * g[j] + b[j];
  v = fmaxf(v, 0.f);
  if (Yf) Yf[i] = v;
  if (Yh) Yh[i] = (_Float16)v;
}

// ---------------- message passing: agg[dst] += z[src] + ee ----------------
__global__ void msg_agg_kernel(const float* __restrict__ z, const float* __restrict__ ee,
                               const int* __restrict__ src, const int* __restrict__ dst,
                               float* __restrict__ agg, size_t total) {
  size_t i = (size_t)blockIdx.x * blockDim.x + threadIdx.x;
  if (i >= total) return;
  const int e = (int)(i >> 9);      // / 512
  const int h = (int)(i & 511);
  const float v = z[(size_t)src[e] * kH + h] + ee[i];
  atomicAdd(&agg[(size_t)dst[e] * kH + h], v);
}

// ---------------- attention head ----------------
__global__ void att_dot_kernel(const float* __restrict__ zg, const float* __restrict__ attW,
                               const float* __restrict__ attB, float* __restrict__ att, int N) {
  int n = blockIdx.x * blockDim.x + threadIdx.x;
  if (n >= N) return;
  float s = attB[0];
  const float* row = zg + (size_t)n * kH;
  for (int j = 0; j < kH; ++j) s += row[j] * attW[j];
  att[n] = s;
}

__global__ void attn_pool_kernel(const float* __restrict__ att, const float* __restrict__ zo,
                                 float* __restrict__ hg, _Float16* __restrict__ hgh) {
  const int b = blockIdx.x;
  __shared__ float sa[kNPG];
  if (threadIdx.x < kNPG) sa[threadIdx.x] = att[b * kNPG + threadIdx.x];
  __syncthreads();
  float m = -1e30f;
  for (int n = 0; n < kNPG; ++n) m = fmaxf(m, sa[n]);
  float denom = 0.f;
  for (int n = 0; n < kNPG; ++n) denom += __expf(sa[n] - m);
  const float inv = 1.f / denom;
  for (int o = threadIdx.x; o < kOUT; o += blockDim.x) {
    float s = 0.f;
    for (int n = 0; n < kNPG; ++n)
      s += __expf(sa[n] - m) * inv * zo[((size_t)b * kNPG + n) * kOUT + o];
    hg[(size_t)b * kOUT + o]  = s;
    hgh[(size_t)b * kOUT + o] = (_Float16)s;
  }
}

// ---------------- host side ----------------
static inline size_t cdiv(size_t a, size_t b) { return (a + b - 1) / b; }

static inline void launch_gemm(const _Float16* A, const _Float16* Bt, const float* bias,
                               float* C, int M, int N, int K, hipStream_t s) {
  const int waves  = (M / 16) * (N / 64);
  const int blocks = (waves + 7) / 8;       // 8 waves per 256-thread block
  gemm_wmma_f16<<<blocks, 256, 0, s>>>(A, Bt, bias, C, M, N, K);
}

static inline void launch_bn(const float* X, int M, int N,
                             float* sums, float* sqs, float* mean, float* rstd,
                             hipStream_t s) {
  zero_f32_kernel<<<(unsigned)cdiv(2 * (size_t)N, 256), 256, 0, s>>>(sums, 2 * (size_t)N);
  dim3 grid(N / 64, 64);
  bn_partial_kernel<<<grid, 256, 0, s>>>(X, M, N, sums, sqs);
  bn_finalize_kernel<<<(unsigned)cdiv((size_t)N, 256), 256, 0, s>>>(sums, sqs, mean, rstd, M, N);
}

extern "C" void kernel_launch(void* const* d_in, const int* in_sizes, int n_in,
                              void* d_out, int out_size, void* d_ws, size_t ws_size,
                              hipStream_t stream) {
  (void)in_sizes; (void)n_in; (void)out_size; (void)ws_size;

  // ---- inputs (setup_inputs order) ----
  const int*   x         = (const int*)  d_in[0];
  const int*   edge_attr = (const int*)  d_in[1];
  const int*   edge_index= (const int*)  d_in[2];
  // d_in[3] = batch (dense, derivable; unused)
  const float* node_tab  = (const float*)d_in[4];
  const float* edge_tab  = (const float*)d_in[5];
  const float* bondW     = (const float*)d_in[6];
  const float* bondB     = (const float*)d_in[7];
  const float* bondG     = (const float*)d_in[8];
  const float* bondBeta  = (const float*)d_in[9];
  const float* atomW1    = (const float*)d_in[10];
  const float* atomB1    = (const float*)d_in[11];
  const float* atomG     = (const float*)d_in[12];
  const float* atomBeta  = (const float*)d_in[13];
  const float* atomW2    = (const float*)d_in[14];
  const float* atomB2    = (const float*)d_in[15];
  const float* normG     = (const float*)d_in[16];
  const float* normB     = (const float*)d_in[17];
  const float* outW      = (const float*)d_in[18];
  const float* outB      = (const float*)d_in[19];
  const float* gapW      = (const float*)d_in[20];
  const float* gapB      = (const float*)d_in[21];
  const float* attW      = (const float*)d_in[22];
  const float* attB      = (const float*)d_in[23];
  const float* projW     = (const float*)d_in[24];
  const float* projB     = (const float*)d_in[25];

  const int* src = edge_index;           // row 0
  const int* dst = edge_index + kE;      // row 1

  // ---- workspace bump allocator (256B aligned) ----
  size_t off = 0;
  auto alloc = [&](size_t bytes) -> void* {
    void* p = (char*)d_ws + off;
    off += (bytes + 255) & ~(size_t)255;
    return p;
  };

  // 256MB region shared by ee (E x H), t1 (N x OUT), zo (N x OUT)
  float*     big    = (float*)    alloc((size_t)kE * kH * 4);
  float*     ee     = big;
  float*     t1     = big;
  float*     zo     = big;
  float*     z_f32  = (float*)    alloc((size_t)kN * kH * 4);
  _Float16*  z_h16  = (_Float16*) alloc((size_t)kN * kH * 2);
  _Float16*  e_h16  = (_Float16*) alloc((size_t)kE * kHE * 2);
  float*     agg    = (float*)    alloc((size_t)kN * kH * 4);
  _Float16*  s_h16  = (_Float16*) alloc((size_t)kN * kH * 2);
  _Float16*  h_h16  = (_Float16*) alloc((size_t)kN * kOUT * 2);
  float*     t2     = (float*)    alloc((size_t)kN * kH * 4);
  _Float16*  zo_h16 = (_Float16*) alloc((size_t)kN * kOUT * 2);
  float*     zg     = (float*)    alloc((size_t)kN * kH * 4);
  float*     att    = (float*)    alloc((size_t)kN * 4);
  float*     hg     = (float*)    alloc((size_t)kB * kOUT * 4);
  _Float16*  hg_h16 = (_Float16*) alloc((size_t)kB * kOUT * 2);
  float*     sums   = (float*)    alloc(2 * (size_t)kOUT * 4);  // sums | sqs contiguous
  float*     sqs    = sums + kOUT;
  float*     mean   = (float*)    alloc((size_t)kOUT * 4);
  float*     rstd   = (float*)    alloc((size_t)kOUT * 4);
  _Float16*  bondWt = (_Float16*) alloc((size_t)kL * kH * kHE * 2);   // [H][HE] per layer
  _Float16*  a1t    = (_Float16*) alloc((size_t)kL * kOUT * kH * 2);  // [OUT][H]
  _Float16*  a2t    = (_Float16*) alloc((size_t)kL * kH * kOUT * 2);  // [H][OUT]
  _Float16*  outWt  = (_Float16*) alloc((size_t)kOUT * kH * 2);
  _Float16*  gapWt  = (_Float16*) alloc((size_t)kH * kOUT * 2);
  _Float16*  projWt = (_Float16*) alloc((size_t)kOUT * kOUT * 2);

  const unsigned TB = 256;
  auto G = [&](size_t n) { return (unsigned)cdiv(n, TB); };

  // ---- weight prep: transpose + f16 convert (deterministic each call) ----
  for (int l = 0; l < kL; ++l) {
    transpose_cvt_kernel<<<G((size_t)kHE * kH), TB, 0, stream>>>(
        bondW + (size_t)l * kHE * kH, bondWt + (size_t)l * kH * kHE, kHE, kH);
    transpose_cvt_kernel<<<G((size_t)kH * kOUT), TB, 0, stream>>>(
        atomW1 + (size_t)l * kH * kOUT, a1t + (size_t)l * kOUT * kH, kH, kOUT);
    transpose_cvt_kernel<<<G((size_t)kOUT * kH), TB, 0, stream>>>(
        atomW2 + (size_t)l * kOUT * kH, a2t + (size_t)l * kH * kOUT, kOUT, kH);
  }
  transpose_cvt_kernel<<<G((size_t)kH * kOUT), TB, 0, stream>>>(outW, outWt, kH, kOUT);
  transpose_cvt_kernel<<<G((size_t)kOUT * kH), TB, 0, stream>>>(gapW, gapWt, kOUT, kH);
  transpose_cvt_kernel<<<G((size_t)kOUT * kOUT), TB, 0, stream>>>(projW, projWt, kOUT, kOUT);

  // ---- embeddings ----
  node_embed_kernel<<<G((size_t)kN * kH), TB, 0, stream>>>(x, node_tab, z_f32, z_h16, (size_t)kN * kH);
  edge_embed_kernel<<<G((size_t)kE * kHE), TB, 0, stream>>>(edge_attr, edge_tab, e_h16, (size_t)kE * kHE);

  // ---- 6 GIN layers ----
  for (int l = 0; l < kL; ++l) {
    // mlp_bond: ee = relu(bn(e_emb @ bW + bB))
    launch_gemm(e_h16, bondWt + (size_t)l * kH * kHE, bondB + (size_t)l * kH,
                ee, kE, kH, kHE, stream);
    launch_bn(ee, kE, kH, sums, sqs, mean, rstd, stream);
    bn_relu_apply_kernel<<<G((size_t)kE * kH), TB, 0, stream>>>(
        ee, mean, rstd, bondG + (size_t)l * kH, bondBeta + (size_t)l * kH,
        ee, (_Float16*)nullptr, (size_t)kE * kH, kH);

    // message + aggregate: agg[dst] += z[src] + ee
    zero_f32_kernel<<<G((size_t)kN * kH), TB, 0, stream>>>(agg, (size_t)kN * kH);
    msg_agg_kernel<<<G((size_t)kE * kH), TB, 0, stream>>>(z_f32, ee, src, dst, agg, (size_t)kE * kH);
    add_cvt_kernel<<<G((size_t)kN * kH), TB, 0, stream>>>(z_f32, agg, s_h16, (size_t)kN * kH);

    // mlp_atom layer 1: h = relu(bn((z+agg) @ aW1 + aB1))
    launch_gemm(s_h16, a1t + (size_t)l * kOUT * kH, atomB1 + (size_t)l * kOUT,
                t1, kN, kOUT, kH, stream);
    launch_bn(t1, kN, kOUT, sums, sqs, mean, rstd, stream);
    bn_relu_apply_kernel<<<G((size_t)kN * kOUT), TB, 0, stream>>>(
        t1, mean, rstd, atomG + (size_t)l * kOUT, atomBeta + (size_t)l * kOUT,
        (float*)nullptr, h_h16, (size_t)kN * kOUT, kOUT);

    // mlp_atom layer 2 + outer BN + ReLU -> new z (f32 + f16)
    launch_gemm(h_h16, a2t + (size_t)l * kH * kOUT, atomB2 + (size_t)l * kH,
                t2, kN, kH, kOUT, stream);
    launch_bn(t2, kN, kH, sums, sqs, mean, rstd, stream);
    bn_relu_apply_kernel<<<G((size_t)kN * kH), TB, 0, stream>>>(
        t2, mean, rstd, normG + (size_t)l * kH, normB + (size_t)l * kH,
        z_f32, z_h16, (size_t)kN * kH, kH);
  }

  // ---- readout ----
  launch_gemm(z_h16, outWt, outB, zo, kN, kOUT, kH, stream);                  // [N, OUT]
  cvt_f16_kernel<<<G((size_t)kN * kOUT), TB, 0, stream>>>(zo, zo_h16, (size_t)kN * kOUT);
  launch_gemm(zo_h16, gapWt, gapB, zg, kN, kH, kOUT, stream);                 // [N, H]
  att_dot_kernel<<<G((size_t)kN), TB, 0, stream>>>(zg, attW, attB, att, kN);  // [N]
  attn_pool_kernel<<<kB, TB, 0, stream>>>(att, zo, hg, hg_h16);               // [B, OUT]
  launch_gemm(hg_h16, projWt, projB, (float*)d_out, kB, kOUT, kOUT, stream);  // [B, OUT]
}
